// Transformer_10015863734272
// MI455X (gfx1250) — matrix operand order repaired and checked by reading, not compiled, not run on hardware
//
#include <hip/hip_runtime.h>
#include <hip/hip_bf16.h>

// ---------------- problem constants ----------------
#define BATCH 32
#define HW    512
#define EMB   256
#define KQD   64
#define VDD   64
#define LAY   6
#define DMLP  128
#define NCLS  10
#define NPAT  1024              // (512/16)^2
#define MTOT  (BATCH*NPAT)      // 32768 rows for all activation GEMMs

typedef unsigned short ushort_t;
typedef __attribute__((ext_vector_type(8)))  short  v8s;
typedef __attribute__((ext_vector_type(16))) short  v16s;
typedef __attribute__((ext_vector_type(8)))  float  v8f;
typedef __attribute__((ext_vector_type(16))) __bf16 v16bf;
typedef __attribute__((ext_vector_type(4)))  unsigned int v4u;
typedef __attribute__((ext_vector_type(8)))  int    v8i;
typedef __attribute__((ext_vector_type(4)))  int    v4i;

// ---------------- bf16 helpers ----------------
__device__ __forceinline__ ushort_t f2bf(float f) {
  unsigned u = __float_as_uint(f);
  u += 0x7FFFu + ((u >> 16) & 1u);          // round-to-nearest-even
  return (ushort_t)(u >> 16);
}
__device__ __forceinline__ float bf2f(ushort_t s) {
  return __uint_as_float(((unsigned)s) << 16);
}

// ---------------- WMMA fragment loaders (16x16x32 bf16; contiguous k: lane half h holds k = 16h + s) ----------------
__device__ __forceinline__ v16bf ld_fragA(const ushort_t* base, int ld) {
  const int lane = threadIdx.x & 31;
  const int lh = lane >> 4, r = lane & 15;
  const ushort_t* p = base + r * ld + lh * 16;
  v8s lo = *(const v8s*)(p);
  v8s hi = *(const v8s*)(p + 8);
  v16s c = __builtin_shufflevector(lo, hi, 0,1,2,3,4,5,6,7,8,9,10,11,12,13,14,15);
  return __builtin_bit_cast(v16bf, c);
}
__device__ __forceinline__ v16bf ld_fragB(const ushort_t* base, int ld) {
  const int lane = threadIdx.x & 31;
  const int lh = lane >> 4, r = lane & 15;
  v16s c = *(const v16s*)(base + r * ld + lh * 16);
  return __builtin_bit_cast(v16bf, c);
}
#define WMMA_BF16(a, b, c) \
  __builtin_amdgcn_wmma_f32_16x16x32_bf16(false, (a), false, (b), (short)0, (c), false, false)

// ---------------- Tensor Data Mover: 2D tile Global -> LDS (ISA 08, D# groups 0/1) -----
#if __has_include(<hip/amd_detail/amd_gfx1250_TDM.h>)
#define TDM_CALL(g0, g1, gz) \
  __builtin_amdgcn_tensor_load_to_lds((g0), (g1), (gz), (gz), (v8i){0,0,0,0,0,0,0,0}, 0)
#else
#define TDM_CALL(g0, g1, gz) \
  __builtin_amdgcn_tensor_load_to_lds((g0), (g1), (gz), (gz), 0)
#endif

// Loads a [tile_d1 rows][tile_d0 elems] bf16 tile; row stride = stride_d0 elements.
__device__ __forceinline__ void tdm_load_2d_bf16(
    unsigned lds_byte_off, const void* gptr,
    unsigned tile_d0, unsigned tile_d1, unsigned stride_d0)
{
  unsigned long long ga = (unsigned long long)gptr;
  v4u g0;
  g0[0] = 1u;                                   // count=1 (valid), user mode
  g0[1] = lds_byte_off;                         // D#.lds_addr
  g0[2] = (unsigned)(ga & 0xFFFFFFFFull);       // global_addr[31:0]
  g0[3] = (unsigned)((ga >> 32) & 0x1FFFFFFull) // global_addr[56:32]
          | (2u << 30);                         // type=2 ("image")
  v8i g1;
  g1[0] = (int)(1u << 16);                      // data_size=1 (2 bytes/elem)
  g1[1] = (int)((tile_d0 & 0xFFFFu) << 16);     // tensor_dim0[15:0] = tile_d0
  g1[2] = (int)((tile_d1 & 0xFFFFu) << 16);     // tensor_dim0 hi=0, tensor_dim1 lo
  g1[3] = (int)((tile_d0 & 0xFFFFu) << 16);     // tensor_dim1 hi=0, tile_dim0
  g1[4] = (int)(tile_d1 & 0xFFFFu);             // tile_dim1; tile_dim2=0 (2D)
  g1[5] = (int)stride_d0;                       // tensor_dim0_stride[31:0]
  g1[6] = 0;                                    // stride hi, dim1_stride lo
  g1[7] = 0;
  v4i gz = {0, 0, 0, 0};
  TDM_CALL(g0, g1, gz);
}

// ---------------- weight convert: src f32 [K][N]  ->  dst bf16 [N][K] ----------------
__global__ void k_wt(const float* __restrict__ src, ushort_t* __restrict__ dst, int K, int N) {
  int i = blockIdx.x * 256 + threadIdx.x;
  if (i < K * N) {
    int n = i / K, k = i - n * K;
    dst[i] = f2bf(src[k * N + n]);
  }
}

// ---------------- patch embed: h = patchify(x) @ Wemb + b + pos  (M=32768,K=256,N=256) ----
__global__ __launch_bounds__(256) void k_patch_embed(
    const float* __restrict__ x, const ushort_t* __restrict__ WembT,
    const float* __restrict__ bemb, const float* __restrict__ pos,
    ushort_t* __restrict__ h)
{
  __shared__ __align__(64) ushort_t As[128 * 32];
  const int tid = threadIdx.x, wave = tid >> 5, lane = tid & 31;
  const int m0 = blockIdx.x * 128;
  const int n0 = blockIdx.y * 64;
  const v8f vz = {0.f,0.f,0.f,0.f,0.f,0.f,0.f,0.f};
  v8f acc[4] = {vz, vz, vz, vz};

  for (int ks = 0; ks < 8; ++ks) {               // K = 256, chunks of 32
    __syncthreads();
    for (int it = 0; it < 16; ++it) {            // stage 128x32 patch values -> bf16
      int lin = tid + it * 256;                  // 0..4095
      int ml = lin >> 5, kk = lin & 31;
      int m = m0 + ml;
      int b = m >> 10, pidx = m & 1023;
      int gy = pidx >> 5, gx = pidx & 31;
      int k = ks * 32 + kk;
      int py = k >> 4, px = k & 15;
      float v = x[((size_t)b * HW + gy * 16 + py) * HW + gx * 16 + px];
      As[ml * 32 + kk] = f2bf(v);
    }
    __syncthreads();
    v16bf aF = ld_fragA(As + wave * 16 * 32, 32);
    v16bf b0 = ld_fragB(WembT + (size_t)(n0 +  0) * EMB + ks * 32, EMB);
    v16bf b1 = ld_fragB(WembT + (size_t)(n0 + 16) * EMB + ks * 32, EMB);
    v16bf b2 = ld_fragB(WembT + (size_t)(n0 + 32) * EMB + ks * 32, EMB);
    v16bf b3 = ld_fragB(WembT + (size_t)(n0 + 48) * EMB + ks * 32, EMB);
    acc[0] = WMMA_BF16(aF, b0, acc[0]);
    acc[1] = WMMA_BF16(aF, b1, acc[1]);
    acc[2] = WMMA_BF16(aF, b2, acc[2]);
    acc[3] = WMMA_BF16(aF, b3, acc[3]);
  }
  const int lh = lane >> 4, r = lane & 15;
  for (int t = 0; t < 4; ++t) {
    int n = n0 + t * 16 + r;
    float bb = bemb[n];
    for (int v = 0; v < 8; ++v) {
      int m = m0 + wave * 16 + v + lh * 8;
      int np = m & 1023;
      h[(size_t)m * EMB + n] = f2bf(acc[t][v] + bb + pos[(size_t)np * EMB + n]);
    }
  }
}

// ---------------- generic GEMM: out[M,N] = act(A_bf16[M,K] @ WtT[N,K] + bias) -------------
// transV!=0: N must be 64; stores transposed per batch: out[b][n][np] (for V^T).
__global__ __launch_bounds__(256) void k_gemm(
    const ushort_t* __restrict__ A, const ushort_t* __restrict__ Wt,
    const float* __restrict__ bias, ushort_t* __restrict__ out,
    int K, int N, int relu, int transV)
{
  const int tid = threadIdx.x, wave = tid >> 5, lane = tid & 31;
  const int m0 = blockIdx.x * 128 + wave * 16;
  const int n0 = blockIdx.y * 64;
  const v8f vz = {0.f,0.f,0.f,0.f,0.f,0.f,0.f,0.f};
  v8f acc[4] = {vz, vz, vz, vz};
  const ushort_t* Ab = A + (size_t)m0 * K;
  const ushort_t* W0 = Wt + (size_t)(n0 +  0) * K;
  const ushort_t* W1 = Wt + (size_t)(n0 + 16) * K;
  const ushort_t* W2 = Wt + (size_t)(n0 + 32) * K;
  const ushort_t* W3 = Wt + (size_t)(n0 + 48) * K;

  v16bf aF = ld_fragA(Ab, K);                    // software pipeline: A one step ahead
  for (int kk = 0; kk < K; kk += 32) {
    __builtin_prefetch(Ab + kk + 128, 0, 1);     // global_prefetch_b8 on streamed A
    v16bf b0 = ld_fragB(W0 + kk, K);
    v16bf b1 = ld_fragB(W1 + kk, K);
    v16bf b2 = ld_fragB(W2 + kk, K);
    v16bf b3 = ld_fragB(W3 + kk, K);
    v16bf aN = (kk + 32 < K) ? ld_fragA(Ab + kk + 32, K) : aF;
    acc[0] = WMMA_BF16(aF, b0, acc[0]);
    acc[1] = WMMA_BF16(aF, b1, acc[1]);
    acc[2] = WMMA_BF16(aF, b2, acc[2]);
    acc[3] = WMMA_BF16(aF, b3, acc[3]);
    aF = aN;
  }
  const int lh = lane >> 4, r = lane & 15;
  if (!transV) {
    for (int t = 0; t < 4; ++t) {
      int n = n0 + t * 16 + r;
      float bb = bias[n];
      for (int v = 0; v < 8; ++v) {
        float xv = acc[t][v] + bb;
        if (relu) xv = fmaxf(xv, 0.f);
        out[(size_t)(m0 + v + lh * 8) * N + n] = f2bf(xv);
      }
    }
  } else {
    int b = m0 >> 10;                            // 128 | 1024, uniform per block
    for (int t = 0; t < 4; ++t) {
      int n = n0 + t * 16 + r;
      float bb = bias[n];
      for (int v = 0; v < 8; ++v) {
        int np = (m0 + v + lh * 8) & 1023;
        out[((size_t)b * VDD + n) * NPAT + np] = f2bf(acc[t][v] + bb);
      }
    }
  }
}

// ---------------- flash attention: per (batch, 128-query tile) ----------------
// Dynamic LDS layout (byte offsets; dynamic-only LDS starts at 0 -> valid D#.lds_addr):
//   Qs :      0 .. 16383    (128 x 64 bf16)
//   Ks0:  16384 .. 32767    (128 x 64 bf16)   double-buffered K chunks
//   Ks1:  32768 .. 49151
//   Vs0:  49152 .. 65535    ( 64 x 128 bf16)  double-buffered V^T chunks
//   Vs1:  65536 .. 81919
//   Ps :  81920 .. 114687   (8 waves x 16 x 128 bf16)
#define ATTN_LDS_BYTES (112 * 1024)
__global__ __launch_bounds__(256) void k_attn(
    const ushort_t* __restrict__ Q, const ushort_t* __restrict__ Kx,
    const ushort_t* __restrict__ Vt, ushort_t* __restrict__ AO)
{
  extern __shared__ __align__(64) ushort_t smem[];
  ushort_t* Qs = smem;                                   // byte 0
  ushort_t* Ps = smem + 40960;                           // byte 81920
  const int tid = threadIdx.x, wave = tid >> 5, lane = tid & 31;
  const int b = blockIdx.y, q0 = blockIdx.x * 128;
  const int lh = lane >> 4, r = lane & 15;
  const v8f vz = {0.f,0.f,0.f,0.f,0.f,0.f,0.f,0.f};

  const ushort_t* Kg = Kx + (size_t)b * NPAT * KQD;      // [key][64]
  const ushort_t* Vg = Vt + (size_t)b * VDD * NPAT;      // [vd][1024]

  // Prologue: wave 0 issues async DMA for Q and chunk 0 (TENSORcnt in-order).
  if (wave == 0) {
    tdm_load_2d_bf16(0u,     Q + ((size_t)b * NPAT + q0) * KQD, KQD, 128, KQD);
    tdm_load_2d_bf16(16384u, Kg,                                KQD, 128, KQD);
    tdm_load_2d_bf16(49152u, Vg,                                128, VDD, NPAT);
  }

  float mrun[8], lrun[8];
  v8f oacc[4] = {vz, vz, vz, vz};
  for (int v = 0; v < 8; ++v) { mrun[v] = -1e30f; lrun[v] = 0.f; }

  for (int kc = 0; kc < 8; ++kc) {                 // 8 key chunks of 128
    if (wave == 0) {
      if (kc < 7) {
        // Prefetch chunk kc+1 into the other buffer, then allow those 2 transfers
        // to stay in flight while guaranteeing chunk kc (and Q) have landed.
        unsigned bi = (unsigned)((kc + 1) & 1);
        tdm_load_2d_bf16(16384u + bi * 16384u, Kg + (size_t)(kc + 1) * 128 * KQD,
                         KQD, 128, KQD);
        tdm_load_2d_bf16(49152u + bi * 16384u, Vg + (size_t)(kc + 1) * 128,
                         128, VDD, NPAT);
        __builtin_amdgcn_s_wait_tensorcnt(2);
      } else {
        __builtin_amdgcn_s_wait_tensorcnt(0);
      }
    }
    __syncthreads();                               // chunk kc visible to all waves
    // Runtime address arithmetic (no pointer-array static initializers from LDS):
    const ushort_t* Kc = smem +  8192 + (kc & 1) * 8192;   // bytes 16384 / 32768
    const ushort_t* Vc = smem + 24576 + (kc & 1) * 8192;   // bytes 49152 / 65536

    // S = Q @ K^T  (16 queries x 128 keys per wave); A frags invariant across t
    v16bf aF0 = ld_fragA(Qs + wave * 16 * 64,      64);
    v16bf aF1 = ld_fragA(Qs + wave * 16 * 64 + 32, 64);
    v8f s[8] = {vz, vz, vz, vz, vz, vz, vz, vz};
    for (int t = 0; t < 8; ++t) {
      v16bf bF0 = ld_fragB(Kc + t * 16 * 64,      64);
      v16bf bF1 = ld_fragB(Kc + t * 16 * 64 + 32, 64);
      s[t] = WMMA_BF16(aF0, bF0, s[t]);
      s[t] = WMMA_BF16(aF1, bF1, s[t]);
    }

    // online softmax (rows live in (VGPR v, lane-half); reduce across 16-lane halves)
    float corr[8];
    for (int v = 0; v < 8; ++v) {
      float cm = -1e30f;
      for (int t = 0; t < 8; ++t) { s[t][v] *= 0.125f; cm = fmaxf(cm, s[t][v]); }
      for (int msk = 1; msk < 16; msk <<= 1) cm = fmaxf(cm, __shfl_xor(cm, msk, 32));
      float mn = fmaxf(mrun[v], cm);
      corr[v] = __expf(mrun[v] - mn);
      float rs = 0.f;
      for (int t = 0; t < 8; ++t) {
        float p = __expf(s[t][v] - mn);
        rs += p;
        Ps[wave * 2048 + (v + lh * 8) * 128 + t * 16 + r] = f2bf(p);
      }
      for (int msk = 1; msk < 16; msk <<= 1) rs += __shfl_xor(rs, msk, 32);
      lrun[v] = lrun[v] * corr[v] + rs;
      mrun[v] = mn;
    }
    for (int t = 0; t < 4; ++t)
      for (int v = 0; v < 8; ++v) oacc[t][v] *= corr[v];

    // O += P @ V   (K = 128 keys -> 4 k-steps; per-wave Ps region, in-order LDS)
    for (int k4 = 0; k4 < 4; ++k4) {
      v16bf aF = ld_fragA(Ps + wave * 2048 + k4 * 32, 128);
      v16bf bF0 = ld_fragB(Vc +  0 * 128 + k4 * 32, 128);
      v16bf bF1 = ld_fragB(Vc + 16 * 128 + k4 * 32, 128);
      v16bf bF2 = ld_fragB(Vc + 32 * 128 + k4 * 32, 128);
      v16bf bF3 = ld_fragB(Vc + 48 * 128 + k4 * 32, 128);
      oacc[0] = WMMA_BF16(aF, bF0, oacc[0]);
      oacc[1] = WMMA_BF16(aF, bF1, oacc[1]);
      oacc[2] = WMMA_BF16(aF, bF2, oacc[2]);
      oacc[3] = WMMA_BF16(aF, bF3, oacc[3]);
    }
    __syncthreads();   // all waves done with buffer (kc&1) before it is re-filled
  }

  for (int t = 0; t < 4; ++t)
    for (int v = 0; v < 8; ++v) {
      int q = q0 + wave * 16 + v + lh * 8;
      AO[((size_t)b * NPAT + q) * VDD + t * 16 + r] = f2bf(oacc[t][v] / lrun[v]);
    }
}

// ---------------- mean pool + classifier (tiny, f32) ----------------
__global__ __launch_bounds__(256) void k_head(
    const ushort_t* __restrict__ h, const float* __restrict__ Wc,
    const float* __restrict__ bc, float* __restrict__ out)
{
  __shared__ float pe[EMB];
  int b = blockIdx.x, e = threadIdx.x;
  float s = 0.f;
  for (int np = 0; np < NPAT; ++np) s += bf2f(h[((size_t)b * NPAT + np) * EMB + e]);
  pe[e] = s * (1.0f / NPAT);
  __syncthreads();
  if (e < NCLS) {
    float a = bc[e];
    for (int k = 0; k < EMB; ++k) a += pe[k] * Wc[k * NCLS + e];
    out[b * NCLS + e] = a;
  }
}

// ---------------- host orchestration ----------------
extern "C" void kernel_launch(void* const* d_in, const int* in_sizes, int n_in,
                              void* d_out, int out_size, void* d_ws, size_t ws_size,
                              hipStream_t stream) {
  const float* x    = (const float*)d_in[0];
  const float* Wemb = (const float*)d_in[1];
  const float* bemb = (const float*)d_in[2];
  const float* pos  = (const float*)d_in[3];
  const float* Wq   = (const float*)d_in[4];
  const float* bq   = (const float*)d_in[5];
  const float* Wk   = (const float*)d_in[6];
  const float* bk   = (const float*)d_in[7];
  const float* Wv   = (const float*)d_in[8];
  const float* bv   = (const float*)d_in[9];
  const float* Wo   = (const float*)d_in[10];
  const float* bo   = (const float*)d_in[11];
  const float* W1   = (const float*)d_in[12];
  const float* b1   = (const float*)d_in[13];
  const float* W2   = (const float*)d_in[14];
  const float* b2   = (const float*)d_in[15];
  const float* Wc   = (const float*)d_in[16];
  const float* bc   = (const float*)d_in[17];

  char* ws = (char*)d_ws;
  size_t off = 0;
  auto alloc = [&](size_t bytes) -> void* {
    void* p = ws + off;
    off += (bytes + 255) & ~(size_t)255;
    return p;
  };
  ushort_t* hA  = (ushort_t*)alloc((size_t)MTOT * EMB * 2);    // 16 MB activations
  ushort_t* oE  = (ushort_t*)alloc((size_t)MTOT * EMB * 2);    // 16 MB attn-proj out
  ushort_t* Qb  = (ushort_t*)alloc((size_t)MTOT * KQD * 2);    // 4 MB
  ushort_t* Kb  = (ushort_t*)alloc((size_t)MTOT * KQD * 2);    // 4 MB
  ushort_t* VtB = (ushort_t*)alloc((size_t)BATCH * VDD * NPAT * 2); // 4 MB (V^T)
  ushort_t* AO  = (ushort_t*)alloc((size_t)MTOT * VDD * 2);    // 4 MB
  ushort_t* F1  = (ushort_t*)alloc((size_t)MTOT * DMLP * 2);   // 8 MB
  ushort_t* WembT = (ushort_t*)alloc((size_t)EMB * EMB * 2);
  ushort_t* WqT = (ushort_t*)alloc((size_t)LAY * KQD * EMB * 2);
  ushort_t* WkT = (ushort_t*)alloc((size_t)LAY * KQD * EMB * 2);
  ushort_t* WvT = (ushort_t*)alloc((size_t)LAY * VDD * EMB * 2);
  ushort_t* WoT = (ushort_t*)alloc((size_t)LAY * EMB * VDD * 2);
  ushort_t* W1T = (ushort_t*)alloc((size_t)LAY * DMLP * EMB * 2);
  ushort_t* W2T = (ushort_t*)alloc((size_t)LAY * EMB * DMLP * 2);
  (void)ws_size; (void)in_sizes; (void)n_in; (void)out_size;

  // ---- weights -> bf16, transposed to [N][K] (L2-resident, ~1.7 MB) ----
  k_wt<<<(EMB * EMB + 255) / 256, 256, 0, stream>>>(Wemb, WembT, EMB, EMB);
  for (int l = 0; l < LAY; ++l) {
    k_wt<<<(EMB * KQD + 255) / 256, 256, 0, stream>>>(Wq + (size_t)l * EMB * KQD, WqT + (size_t)l * KQD * EMB, EMB, KQD);
    k_wt<<<(EMB * KQD + 255) / 256, 256, 0, stream>>>(Wk + (size_t)l * EMB * KQD, WkT + (size_t)l * KQD * EMB, EMB, KQD);
    k_wt<<<(EMB * VDD + 255) / 256, 256, 0, stream>>>(Wv + (size_t)l * EMB * VDD, WvT + (size_t)l * VDD * EMB, EMB, VDD);
    k_wt<<<(VDD * EMB + 255) / 256, 256, 0, stream>>>(Wo + (size_t)l * VDD * EMB, WoT + (size_t)l * EMB * VDD, VDD, EMB);
    k_wt<<<(EMB * DMLP + 255) / 256, 256, 0, stream>>>(W1 + (size_t)l * EMB * DMLP, W1T + (size_t)l * DMLP * EMB, EMB, DMLP);
    k_wt<<<(DMLP * EMB + 255) / 256, 256, 0, stream>>>(W2 + (size_t)l * DMLP * EMB, W2T + (size_t)l * EMB * DMLP, DMLP, EMB);
  }

  // ---- patch embed ----
  k_patch_embed<<<dim3(MTOT / 128, EMB / 64), 256, 0, stream>>>(x, WembT, bemb, pos, hA);

  // ---- 6 transformer layers ----
  for (int l = 0; l < LAY; ++l) {
    k_gemm<<<dim3(MTOT / 128, 1), 256, 0, stream>>>(hA, WqT + (size_t)l * KQD * EMB, bq + l * KQD, Qb, EMB, KQD, 0, 0);
    k_gemm<<<dim3(MTOT / 128, 1), 256, 0, stream>>>(hA, WkT + (size_t)l * KQD * EMB, bk + l * KQD, Kb, EMB, KQD, 0, 0);
    k_gemm<<<dim3(MTOT / 128, 1), 256, 0, stream>>>(hA, WvT + (size_t)l * VDD * EMB, bv + l * VDD, VtB, EMB, VDD, 0, 1);
    k_attn<<<dim3(NPAT / 128, BATCH), 256, ATTN_LDS_BYTES, stream>>>(Qb, Kb, VtB, AO);
    k_gemm<<<dim3(MTOT / 128, EMB / 64), 256, 0, stream>>>(AO, WoT + (size_t)l * EMB * VDD, bo + l * EMB, oE, VDD, EMB, 0, 0);
    k_gemm<<<dim3(MTOT / 128, DMLP / 64), 256, 0, stream>>>(oE, W1T + (size_t)l * DMLP * EMB, b1 + l * DMLP, F1, EMB, DMLP, 1, 0);
    k_gemm<<<dim3(MTOT / 128, EMB / 64), 256, 0, stream>>>(F1, W2T + (size_t)l * EMB * DMLP, b2 + l * EMB, hA, DMLP, EMB, 0, 0);
  }

  // ---- pool + classify ----
  k_head<<<BATCH, 256, 0, stream>>>(hA, Wc, bc, (float*)d_out);
}
